// LSTM_1400159338669
// MI455X (gfx1250) — matrix operand order repaired and checked by reading, not compiled, not run on hardware
//
#include <hip/hip_runtime.h>
#include <stdint.h>
#include <stddef.h>

// ---------------- problem dims (compile-time) ----------------
#define T_STEPS 512
#define BATCH   64
#define IN_DIM  512
#define HID     1024
#define GATES   (4*HID)

typedef __attribute__((ext_vector_type(16))) __bf16 v16bf;
typedef __attribute__((ext_vector_type(8)))  float  v8f;

union Frag {
  unsigned u[8];
  uint4    q[2];
  v16bf    v;
};

// ---------------- bf16 split helpers (hi + lo ~= f32) ----------------
__device__ __forceinline__ unsigned short bf16_rne_bits(float f) {
  unsigned u = __builtin_bit_cast(unsigned, f);
  u += 0x7FFFu + ((u >> 16) & 1u);                // round-to-nearest-even
  return (unsigned short)(u >> 16);
}
__device__ __forceinline__ float bf16_bits_to_f(unsigned short b) {
  unsigned u = ((unsigned)b) << 16;
  return __builtin_bit_cast(float, u);
}
__device__ __forceinline__ void split2(float f0, float f1, unsigned &hi, unsigned &lo) {
  unsigned short h0 = bf16_rne_bits(f0);
  unsigned short h1 = bf16_rne_bits(f1);
  unsigned short l0 = bf16_rne_bits(f0 - bf16_bits_to_f(h0));
  unsigned short l1 = bf16_rne_bits(f1 - bf16_bits_to_f(h1));
  hi = (unsigned)h0 | ((unsigned)h1 << 16);
  lo = (unsigned)l0 | ((unsigned)l1 << 16);
}

__device__ __forceinline__ float sigmoid_f(float x) {
  return 1.0f / (1.0f + __expf(-x));
}
__device__ __forceinline__ float tanh_f(float x) {
  float ax = fabsf(x);
  float e  = __expf(-2.0f * ax);
  float t  = (1.0f - e) / (1.0f + e);
  return copysignf(t, x);
}

// ---------------- B-fragment packing (weights) ----------------
// B (KxN) tile 32x16 bf16, per ISA: N=lane&15; dword v:
// k = (v<4 ? 2v : 2v+8) + 8*(lane>=16). Storage: tile*256 + lane*8 + v dwords.
__global__ void pack_w_kernel(const float* __restrict__ W, int Kdim, int numKt,
                              unsigned* __restrict__ hi, unsigned* __restrict__ lo,
                              int ndwords) {
  int idx = blockIdx.x * 256 + threadIdx.x;
  if (idx >= ndwords) return;
  int tile = idx >> 8;
  int rem  = idx & 255;
  int lane = rem >> 3;
  int v    = rem & 7;
  int nt = tile / numKt;
  int kt = tile - nt * numKt;
  int n  = nt * 16 + (lane & 15);
  int k  = kt * 32 + ((v < 4) ? 2 * v : 2 * v + 8) + ((lane & 16) ? 8 : 0);
  const float* p = W + (size_t)n * Kdim + k;     // W is [N][K] row-major
  unsigned h, l;
  split2(p[0], p[1], h, l);
  hi[idx] = h;
  lo[idx] = l;
}

// ---------------- A-fragment packing (x, h0) ----------------
// A (MxK) tile 16x32 bf16, per ISA: lane = M row (l&15); dword v:
// k = (v<4 ? 2v : 2v+8) + 8*(lane>=16). Storage: tile*256 + lane*8 + v dwords.
__global__ void pack_a_kernel(const float* __restrict__ A, int Kdim, int numKt,
                              unsigned* __restrict__ hi, unsigned* __restrict__ lo,
                              int ndwords) {
  int idx = blockIdx.x * 256 + threadIdx.x;
  if (idx >= ndwords) return;
  int tile = idx >> 8;
  int rem  = idx & 255;
  int lane = rem >> 3;
  int v    = rem & 7;
  int mt = tile / numKt;
  int kt = tile - mt * numKt;
  int m  = mt * 16 + (lane & 15);
  int k  = kt * 32 + ((v < 4) ? 2 * v : 2 * v + 8) + ((lane & 16) ? 8 : 0);
  const float* p = A + (size_t)m * Kdim + k;
  unsigned h, l;
  split2(p[0], p[1], h, l);
  hi[idx] = h;
  lo[idx] = l;
}

__global__ void bar_reset_kernel(unsigned* __restrict__ bar) {
  if (threadIdx.x < 2) bar[threadIdx.x] = 0u;
}

// ---------------- phase 1: gates_x = x . W_ih^T + (b_ih + b_hh) ----------------
// One wave per (16-row M-tile, 4 consecutive 16-col N-tiles). A and B both come
// from pre-packed bf16 hi/lo fragments -> inner loop is b128 loads + WMMA only.
__global__ __launch_bounds__(256) void gemm_gates_x_kernel(
    const unsigned* __restrict__ xp_hi, const unsigned* __restrict__ xp_lo,
    const unsigned* __restrict__ wih_hi, const unsigned* __restrict__ wih_lo,
    const float* __restrict__ b_ih, const float* __restrict__ b_hh,
    float* __restrict__ gx) {
  const int lane = threadIdx.x & 31;
  const int wgl  = blockIdx.x * 8 + (threadIdx.x >> 5);   // 0..131071
  const int mt   = wgl >> 6;                              // 0..2047  (T*B/16)
  const int s    = wgl & 63;                              // N strip (4 tiles each)
  const int lm   = lane & 15;
  const int lhi8 = (lane & 16) ? 8 : 0;
  const int NKT  = IN_DIM / 32;                           // 16

  v8f acc[4];
#pragma unroll
  for (int g = 0; g < 4; ++g)
#pragma unroll
    for (int r = 0; r < 8; ++r) acc[g][r] = 0.0f;

  for (int kt = 0; kt < NKT; ++kt) {
    size_t at = ((size_t)(mt * NKT + kt)) * 256 + lane * 8;
    Frag ah, al;
    ah.q[0] = *(const uint4*)(xp_hi + at);
    ah.q[1] = *(const uint4*)(xp_hi + at + 4);
    al.q[0] = *(const uint4*)(xp_lo + at);
    al.q[1] = *(const uint4*)(xp_lo + at + 4);
#pragma unroll
    for (int ni = 0; ni < 4; ++ni) {
      size_t tb = ((size_t)(s * 4 + ni) * NKT + kt) * 256 + lane * 8;
      Frag bh, bl;
      bh.q[0] = *(const uint4*)(wih_hi + tb);
      bh.q[1] = *(const uint4*)(wih_hi + tb + 4);
      bl.q[0] = *(const uint4*)(wih_lo + tb);
      bl.q[1] = *(const uint4*)(wih_lo + tb + 4);
      acc[ni] = __builtin_amdgcn_wmma_f32_16x16x32_bf16(false, ah.v, false, bh.v, (short)0, acc[ni], false, false);
      acc[ni] = __builtin_amdgcn_wmma_f32_16x16x32_bf16(false, ah.v, false, bl.v, (short)0, acc[ni], false, false);
      acc[ni] = __builtin_amdgcn_wmma_f32_16x16x32_bf16(false, al.v, false, bh.v, (short)0, acc[ni], false, false);
    }
  }
#pragma unroll
  for (int ni = 0; ni < 4; ++ni) {
    int n = (s * 4 + ni) * 16 + lm;
    float bias = b_ih[n] + b_hh[n];
#pragma unroll
    for (int r = 0; r < 8; ++r) {
      int row = mt * 16 + r + lhi8;                 // C layout: m = r + 8*(lane>=16)
      gx[(size_t)row * GATES + n] = acc[ni][r] + bias;
    }
  }
}

// ---------------- grid-wide barrier (32 co-resident blocks) ----------------
__device__ __forceinline__ void grid_sync(volatile unsigned* cnt, volatile unsigned* gen,
                                          unsigned nblocks) {
  __syncthreads();
  if (threadIdx.x == 0) {
    __threadfence();
    unsigned g = *gen;
    unsigned prev = atomicAdd((unsigned*)cnt, 1u);
    if (prev == nblocks - 1u) {
      *cnt = 0u;
      __threadfence();
      atomicAdd((unsigned*)gen, 1u);
    } else {
      while (*gen == g) { __builtin_amdgcn_s_sleep(2); }
    }
    __threadfence();
  }
  __syncthreads();
}

// ---------------- phase 2: persistent recurrent kernel ----------------
// 32 blocks x 256 threads = 256 waves. Wave (mt 0..3, jt 0..63) owns the 16x16
// tile of ALL FOUR gates for batch rows mt*16..+15 and hidden cols jt*16..+15.
// c lives in registers for the entire sequence. h circulates ONLY as packed
// bf16 hi/lo A-fragments (double-buffered): each wave's h tile is half an
// A-tile (jt even -> dwords v0..3, jt odd -> v4..7 of tile kt=jt>>1), repacked
// through a wave-private LDS tile (in-order DS ops, no block barrier needed).
__global__ __launch_bounds__(256, 1) void lstm_persistent_kernel(
    const unsigned* __restrict__ whh_hi, const unsigned* __restrict__ whh_lo,
    const float* __restrict__ gx, const float* __restrict__ c0,
    unsigned* __restrict__ hA_hi, unsigned* __restrict__ hA_lo,
    float* __restrict__ out, unsigned* bar) {
  const int lane = threadIdx.x & 31;
  const int wave = threadIdx.x >> 5;
  const int wgl  = blockIdx.x * 8 + wave;                 // 0..255
  const int mt   = wgl & 3;
  const int jt   = wgl >> 2;                              // 0..63
  const int lm   = lane & 15;
  const int lhi8 = (lane & 16) ? 8 : 0;
  const int NKT  = HID / 32;                              // 32
  const int HADW = BATCH * HID / 2;                       // 32768 dwords per buffer

  __shared__ float lds_t[8][16 * 18];                     // wave-private 16x16 tile, stride 18
  float* tl = lds_t[wave];

  volatile unsigned* bar_cnt = bar;
  volatile unsigned* bar_gen = bar + 1;

  float c_reg[8];
#pragma unroll
  for (int r = 0; r < 8; ++r) {
    int b = mt * 16 + r + lhi8;
    int j = jt * 16 + lm;
    c_reg[r] = c0[(size_t)b * HID + j];
  }

  size_t btile[4];
#pragma unroll
  for (int g = 0; g < 4; ++g)
    btile[g] = (size_t)(g * 64 + jt) * NKT * 256;         // nt = g*64 + jt

  // where this wave writes its packed h half-tile
  const int    atile = mt * NKT + (jt >> 1);
  const int    vbase = (jt & 1) * 4;
  const size_t woff  = (size_t)atile * 256 + lane * 8 + vbase;

  for (int t = 0; t < T_STEPS; ++t) {
    const unsigned* hAh_r = hA_hi + (size_t)(t & 1) * HADW;
    const unsigned* hAl_r = hA_lo + (size_t)(t & 1) * HADW;
    unsigned*       hAh_w = hA_hi + (size_t)((t + 1) & 1) * HADW;
    unsigned*       hAl_w = hA_lo + (size_t)((t + 1) & 1) * HADW;

    v8f acc[4];
#pragma unroll
    for (int g = 0; g < 4; ++g)
#pragma unroll
      for (int r = 0; r < 8; ++r) acc[g][r] = 0.0f;

    for (int kt = 0; kt < NKT; ++kt) {
      size_t at = ((size_t)(mt * NKT + kt)) * 256 + lane * 8;
      Frag ah, al;
      ah.q[0] = *(const uint4*)(hAh_r + at);
      ah.q[1] = *(const uint4*)(hAh_r + at + 4);
      al.q[0] = *(const uint4*)(hAl_r + at);
      al.q[1] = *(const uint4*)(hAl_r + at + 4);
#pragma unroll
      for (int g = 0; g < 4; ++g) {
        size_t tb = btile[g] + (size_t)kt * 256 + lane * 8;
        Frag bh, bl;
        bh.q[0] = *(const uint4*)(whh_hi + tb);
        bh.q[1] = *(const uint4*)(whh_hi + tb + 4);
        bl.q[0] = *(const uint4*)(whh_lo + tb);
        bl.q[1] = *(const uint4*)(whh_lo + tb + 4);
        acc[g] = __builtin_amdgcn_wmma_f32_16x16x32_bf16(false, ah.v, false, bh.v, (short)0, acc[g], false, false);
        acc[g] = __builtin_amdgcn_wmma_f32_16x16x32_bf16(false, ah.v, false, bl.v, (short)0, acc[g], false, false);
        acc[g] = __builtin_amdgcn_wmma_f32_16x16x32_bf16(false, al.v, false, bh.v, (short)0, acc[g], false, false);
      }
    }

    const size_t grow = (size_t)t * BATCH * GATES;
#pragma unroll
    for (int r = 0; r < 8; ++r) {
      int b = mt * 16 + r + lhi8;
      int j = jt * 16 + lm;
      const float* gp = gx + grow + (size_t)b * GATES + jt * 16 + lm;
      float ig = sigmoid_f(acc[0][r] + gp[0 * HID]);
      float fg = sigmoid_f(acc[1][r] + gp[1 * HID]);
      float gg = tanh_f  (acc[2][r] + gp[2 * HID]);
      float og = sigmoid_f(acc[3][r] + gp[3 * HID]);
      float cn = fg * c_reg[r] + ig * gg;
      c_reg[r] = cn;
      float hn = og * tanh_f(cn);
      out[((size_t)t * BATCH + b) * HID + j] = hn;
      tl[(r + lhi8) * 18 + lm] = hn;                      // stage for repack
      if (t == T_STEPS - 1) {
        size_t base = (size_t)T_STEPS * BATCH * HID;
        out[base + (size_t)b * HID + j] = hn;                          // h_T
        out[base + (size_t)BATCH * HID + (size_t)b * HID + j] = cn;    // c_T
      }
    }

    // repack this wave's h tile into A-fragment layout (wave-private LDS,
    // DS ops are in-order within a wave -> no block barrier required)
    {
      uint4 qh, ql;
      unsigned hh[4], ll[4];
#pragma unroll
      for (int i = 0; i < 4; ++i) {
        int jloc = 2 * i + ((lane & 16) ? 8 : 0);
        float2 p = *(const float2*)(tl + lm * 18 + jloc); // stride 18 keeps 8B align
        split2(p.x, p.y, hh[i], ll[i]);
      }
      qh.x = hh[0]; qh.y = hh[1]; qh.z = hh[2]; qh.w = hh[3];
      ql.x = ll[0]; ql.y = ll[1]; ql.z = ll[2]; ql.w = ll[3];
      *(uint4*)(hAh_w + woff) = qh;
      *(uint4*)(hAl_w + woff) = ql;
    }

    grid_sync(bar_cnt, bar_gen, gridDim.x);
  }
}

// ---------------- host launch ----------------
extern "C" void kernel_launch(void* const* d_in, const int* in_sizes, int n_in,
                              void* d_out, int out_size, void* d_ws, size_t ws_size,
                              hipStream_t stream) {
  const float* x    = (const float*)d_in[0];
  const float* w_ih = (const float*)d_in[1];
  const float* w_hh = (const float*)d_in[2];
  const float* b_ih = (const float*)d_in[3];
  const float* b_hh = (const float*)d_in[4];
  const float* h0   = (const float*)d_in[5];
  const float* c0   = (const float*)d_in[6];
  float* out = (float*)d_out;

  // workspace layout: whh 8+8 MB, wih 4+4 MB, x packed 32+32 MB,
  // gates_x 512 MB, packed h 2x(128+128) KB, barrier 8 B  (~601 MB)
  char* ws = (char*)d_ws;
  const size_t WHH_DW = (size_t)GATES * HID / 2;        // 2,097,152
  const size_t WIH_DW = (size_t)GATES * IN_DIM / 2;     // 1,048,576
  const size_t XP_DW  = (size_t)T_STEPS * BATCH * IN_DIM / 2;  // 8,388,608
  const size_t HA_DW  = (size_t)BATCH * HID / 2;        // 32,768 per buffer
  unsigned* whh_hi = (unsigned*)ws;
  unsigned* whh_lo = whh_hi + WHH_DW;
  unsigned* wih_hi = whh_lo + WHH_DW;
  unsigned* wih_lo = wih_hi + WIH_DW;
  unsigned* xp_hi  = wih_lo + WIH_DW;
  unsigned* xp_lo  = xp_hi + XP_DW;
  float*    gx     = (float*)(xp_lo + XP_DW);
  unsigned* hA_hi  = (unsigned*)(gx + (size_t)T_STEPS * BATCH * GATES);
  unsigned* hA_lo  = hA_hi + 2 * HA_DW;
  unsigned* bar    = hA_lo + 2 * HA_DW;

  pack_w_kernel<<<(int)(WHH_DW / 256), 256, 0, stream>>>(w_hh, HID, HID / 32,
                                                         whh_hi, whh_lo, (int)WHH_DW);
  pack_w_kernel<<<(int)(WIH_DW / 256), 256, 0, stream>>>(w_ih, IN_DIM, IN_DIM / 32,
                                                         wih_hi, wih_lo, (int)WIH_DW);
  pack_a_kernel<<<(int)(XP_DW / 256), 256, 0, stream>>>(x, IN_DIM, IN_DIM / 32,
                                                        xp_hi, xp_lo, (int)XP_DW);
  pack_a_kernel<<<(int)(HA_DW / 256), 256, 0, stream>>>(h0, HID, HID / 32,
                                                        hA_hi, hA_lo, (int)HA_DW);
  bar_reset_kernel<<<1, 32, 0, stream>>>(bar);
  gemm_gates_x_kernel<<<(T_STEPS * BATCH / 16) * 64 / 8, 256, 0, stream>>>(
      xp_hi, xp_lo, wih_hi, wih_lo, b_ih, b_hh, gx);
  lstm_persistent_kernel<<<32, 256, 0, stream>>>(whh_hi, whh_lo, gx, c0,
                                                 hA_hi, hA_lo, out, bar);
}
